// VRNN_85667417686554
// MI455X (gfx1250) — matrix operand order repaired and checked by reading, not compiled, not run on hardware
//
#include <hip/hip_runtime.h>
#include <math.h>

typedef __bf16 bf16_t;
typedef __attribute__((ext_vector_type(16))) __bf16 v16bf;
typedef __attribute__((ext_vector_type(8)))  __bf16 v8bf;
typedef __attribute__((ext_vector_type(8)))  float  v8f;

#define B_  128
#define T_  512
#define F_  128
#define H_  256
#define Z_  128

// gfx1250 has native f32->bf16 converts; fptrunc lowers to v_cvt (RNE).
__device__ __forceinline__ bf16_t f2bf(float f) { return static_cast<bf16_t>(f); }

#if __has_builtin(__builtin_amdgcn_tanhf)
__device__ __forceinline__ float tanh_(float x) { return __builtin_amdgcn_tanhf(x); }
#else
__device__ __forceinline__ float tanh_(float x) { return tanhf(x); }
#endif

__device__ __forceinline__ float sigmoidf_(float x) { return 1.f / (1.f + __expf(-x)); }
__device__ __forceinline__ float softplusf_(float x) { return (x > 20.f) ? x : log1pf(__expf(x)); }

// ---- WMMA fragment loaders -------------------------------------------------
// A: 16x32 bf16 tile from row-major [16][lda] LDS buffer.
// lanes 0-15: M=lane, halves 0..7 = K k0..k0+7,  halves 8..15 = K k0+16..k0+23
// lanes 16-31: M=lane-16, K ranges shifted by +8.
__device__ __forceinline__ v16bf ldsA(const bf16_t* buf, int lda, int k0, int lane) {
  int row  = lane & 15;
  int koff = k0 + ((lane & 16) ? 8 : 0);
  const bf16_t* p = buf + row * lda + koff;
  v8bf lo = *reinterpret_cast<const v8bf*>(p);
  v8bf hi = *reinterpret_cast<const v8bf*>(p + 16);
  return __builtin_shufflevector(lo, hi, 0,1,2,3,4,5,6,7,8,9,10,11,12,13,14,15);
}

// B: 32x16 bf16 tile from pre-transposed weights Wt[N][ldk] in global (L2-hot).
// lanes 0-15: N=lane, K k0..k0+7 and k0+16..k0+23; lanes 16-31: K ranges shifted by +8.
__device__ __forceinline__ v16bf gloB(const bf16_t* __restrict__ wT, int ldk,
                                      int n0, int k0, int lane) {
  int n    = n0 + (lane & 15);
  int koff = k0 + ((lane & 16) ? 8 : 0);
  const bf16_t* p = wT + (size_t)n * ldk + koff;
  v8bf lo = *reinterpret_cast<const v8bf*>(p);
  v8bf hi = *reinterpret_cast<const v8bf*>(p + 16);
  return __builtin_shufflevector(lo, hi, 0,1,2,3,4,5,6,7,8,9,10,11,12,13,14,15);
}

__device__ __forceinline__ v8f mmA(const bf16_t* A, int lda,
                                   const bf16_t* __restrict__ wT, int ldk,
                                   int n0, int kTiles, int aK0, int bK0,
                                   v8f acc, int lane) {
#pragma unroll
  for (int kt = 0; kt < kTiles; ++kt) {
    v16bf a = ldsA(A, lda, aK0 + kt * 32, lane);
    v16bf b = gloB(wT, ldk, n0, bK0 + kt * 32, lane);
    acc = __builtin_amdgcn_wmma_f32_16x16x32_bf16(false, a, false, b,
                                                  (short)0, acc, false, false);
  }
  return acc;
}

// D-frag element (m,n): m = r + (lane>=16 ? 8 : 0), n = lane&15 (+n0)
__device__ __forceinline__ void dstore_lds_bf(bf16_t* dst, int ldn, int n0,
                                              const v8f& d, const float* __restrict__ bias,
                                              bool relu, int lane) {
  int n  = n0 + (lane & 15);
  int mb = (lane & 16) ? 8 : 0;
  float bv = bias[n];
#pragma unroll
  for (int r = 0; r < 8; ++r) {
    float v = d[r] + bv;
    if (relu) v = fmaxf(v, 0.f);
    dst[(mb + r) * ldn + n] = f2bf(v);
  }
}

__device__ __forceinline__ void dstore_gmem(float* __restrict__ out, int b0, int t, int n0,
                                            const v8f& d, const float* __restrict__ bias,
                                            bool sp, int lane) {
  int n  = n0 + (lane & 15);
  int mb = (lane & 16) ? 8 : 0;
  float bv = bias[n];
#pragma unroll
  for (int r = 0; r < 8; ++r) {
    float v = d[r] + bv;
    if (sp) v = softplusf_(v);
    out[((size_t)(b0 + mb + r) * T_ + t) * 128 + n] = v;
  }
}

// ---- weight transpose+convert: W[K][N] f32 -> Wt[N][K] bf16 ---------------
__global__ void wconv_kernel(const float* __restrict__ src, bf16_t* __restrict__ dst,
                             int K, int N) {
  int idx = blockIdx.x * blockDim.x + threadIdx.x;
  int total = K * N;
  if (idx < total) {
    int n = idx / K;
    int k = idx - n * K;
    dst[idx] = f2bf(src[k * N + n]);
  }
}

struct VrnnParams {
  const float *x, *eps;
  const float *prior_h_b, *prior_mu_b, *prior_sigma_b, *phi_x_b;
  const float *enc_h_b, *enc_mu_b, *enc_sigma_b, *phi_z_b;
  const float *dec_h_b, *dec_mu_b, *dec_sigma_b, *lstm_bias;
  const bf16_t *phi_xT, *prior_hT, *prior_muT, *prior_sgT;
  const bf16_t *enc_hT, *enc_muT, *enc_sgT, *phi_zT;
  const bf16_t *dec_hT, *dec_muT, *dec_sgT, *lstm_kT, *lstm_recT;
  float *o_rnn, *o_encmu, *o_encsg, *o_decmu, *o_decsg, *o_primu, *o_prisg;
};

// One workgroup = 16 batch rows, 8 waves, full T-step recurrence in-kernel.
__launch_bounds__(256, 1)
__global__ void vrnn_step_kernel(VrnnParams P) {
  __shared__ __align__(16) bf16_t s_hb[16 * 256];      // h state (bf16, matmul A)
  __shared__ float               s_c [16 * 256];       // c state (f32)
  __shared__ __align__(16) bf16_t s_xt[2][16 * 128];   // double-buffered x_t
  __shared__ __align__(16) bf16_t s_x1[16 * 128];
  __shared__ __align__(16) bf16_t s_ph[16 * 128];      // prior_h
  __shared__ __align__(16) bf16_t s_eh[16 * 128];      // enc_h
  __shared__ __align__(16) bf16_t s_z [16 * 128];      // z
  __shared__ __align__(16) bf16_t s_z1[16 * 128];      // z1
  __shared__ __align__(16) bf16_t s_dh[16 * 128];      // dec_h

  const int tid  = threadIdx.x;
  const int lane = tid & 31;
  const int w    = tid >> 5;          // wave index 0..7
  const int b0   = blockIdx.x * 16;   // batch-row base

  for (int i = tid; i < 16 * 256; i += 256) { s_hb[i] = f2bf(0.f); s_c[i] = 0.f; }
  // preload x_0
  for (int i = tid; i < 16 * 128; i += 256) {
    int m = i >> 7, f = i & 127;
    s_xt[0][i] = f2bf(P.x[((size_t)(b0 + m) * T_) * F_ + f]);
  }
  __syncthreads();

  for (int t = 0; t < T_; ++t) {
    const bf16_t* xt  = s_xt[t & 1];
    bf16_t*       xtn = s_xt[(t & 1) ^ 1];

    // P1: x1 = x@phi_x + b (linear) ; prior_h = relu(h@prior_h_w + b)
    { v8f acc = {}; acc = mmA(xt,   128, P.phi_xT, 128, w * 16, 4, 0, 0, acc, lane);
      dstore_lds_bf(s_x1, 128, w * 16, acc, P.phi_x_b, false, lane); }
    { v8f acc = {}; acc = mmA(s_hb, 256, P.prior_hT, 256, w * 16, 8, 0, 0, acc, lane);
      dstore_lds_bf(s_ph, 128, w * 16, acc, P.prior_h_b, true, lane); }
    __syncthreads();

    // P2: enc_h = relu([x1|h]@enc_h_w + b); prior_mu/prior_sigma -> global
    { v8f acc = {};
      acc = mmA(s_x1, 128, P.enc_hT, 384, w * 16, 4, 0,   0, acc, lane);
      acc = mmA(s_hb, 256, P.enc_hT, 384, w * 16, 8, 0, 128, acc, lane);
      dstore_lds_bf(s_eh, 128, w * 16, acc, P.enc_h_b, true, lane); }
    { v8f acc = {}; acc = mmA(s_ph, 128, P.prior_muT, 128, w * 16, 4, 0, 0, acc, lane);
      dstore_gmem(P.o_primu, b0, t, w * 16, acc, P.prior_mu_b, false, lane); }
    { v8f acc = {}; acc = mmA(s_ph, 128, P.prior_sgT, 128, w * 16, 4, 0, 0, acc, lane);
      dstore_gmem(P.o_prisg, b0, t, w * 16, acc, P.prior_sigma_b, true, lane); }
    __syncthreads();

    // P3: enc_mu, enc_sigma, reparameterized z (fused in-register)
    { v8f mu = {}, sg = {};
      mu = mmA(s_eh, 128, P.enc_muT, 128, w * 16, 4, 0, 0, mu, lane);
      sg = mmA(s_eh, 128, P.enc_sgT, 128, w * 16, 4, 0, 0, sg, lane);
      int n  = w * 16 + (lane & 15);
      int mb = (lane & 16) ? 8 : 0;
      float bmu = P.enc_mu_b[n], bsg = P.enc_sigma_b[n];
#pragma unroll
      for (int r = 0; r < 8; ++r) {
        int m = mb + r;
        size_t off = ((size_t)(b0 + m) * T_ + t) * Z_ + n;
        float muv = mu[r] + bmu;
        float sgv = softplusf_(sg[r] + bsg);
        P.o_encmu[off] = muv;
        P.o_encsg[off] = sgv;
        s_z[m * 128 + n] = f2bf(sgv * P.eps[off] + muv);
      }
    }
    __syncthreads();

    // P4: z1 = relu(z@phi_z + b); last step also emits rnn_output
    { v8f acc = {}; acc = mmA(s_z, 128, P.phi_zT, 128, w * 16, 4, 0, 0, acc, lane);
      int n  = w * 16 + (lane & 15);
      int mb = (lane & 16) ? 8 : 0;
      float bv = P.phi_z_b[n];
#pragma unroll
      for (int r = 0; r < 8; ++r) {
        float v = fmaxf(acc[r] + bv, 0.f);
        s_z1[(mb + r) * 128 + n] = f2bf(v);
        if (t == T_ - 1) P.o_rnn[(size_t)(b0 + mb + r) * Z_ + n] = v;
      }
    }
    __syncthreads();

    // P5: dec_h = relu([z1|h]@dec_h_w + b), then LSTM gates in TWO passes of
    // 4 accumulators each (quadrants i/f/g/o for one state tile), finishing
    // the elementwise LSTM per pass; h_next held in registers across barrier.
    { v8f acc = {};
      acc = mmA(s_z1, 128, P.dec_hT, 384, w * 16, 4, 0,   0, acc, lane);
      acc = mmA(s_hb, 256, P.dec_hT, 384, w * 16, 8, 0, 128, acc, lane);
      dstore_lds_bf(s_dh, 128, w * 16, acc, P.dec_h_b, true, lane); }

    // prefetch next step's x tile into the alternate LDS buffer
    if (t + 1 < T_) {
      for (int i = tid; i < 16 * 128; i += 256) {
        int m = i >> 7, f = i & 127;
        xtn[i] = f2bf(P.x[((size_t)(b0 + m) * T_ + (t + 1)) * F_ + f]);
      }
      // warm L2/L0 for next step's eps rows
      if (lane < 16)
        __builtin_prefetch(&P.eps[((size_t)(b0 + lane) * T_ + (t + 1)) * Z_ + w * 16], 0, 1);
    }

    float hn_keep[2][8];
#pragma unroll
    for (int p = 0; p < 2; ++p) {
      const int jt = w + 8 * p;          // state column tile
      v8f qa[4];                          // i, f, g, o accumulators
#pragma unroll
      for (int q = 0; q < 4; ++q) {
        int n0 = (jt + 16 * q) * 16;     // gate tile = quadrant q, columns jt
        v8f acc = {};
        acc = mmA(s_x1, 128, P.lstm_kT,   256, n0, 4, 0,   0, acc, lane);
        acc = mmA(s_z1, 128, P.lstm_kT,   256, n0, 4, 0, 128, acc, lane);
        acc = mmA(s_hb, 256, P.lstm_recT, 256, n0, 8, 0,   0, acc, lane);
        qa[q] = acc;
      }
      int n  = jt * 16 + (lane & 15);
      int mb = (lane & 16) ? 8 : 0;
      float bi = P.lstm_bias[n];
      float bf = P.lstm_bias[256 + n];
      float bg = P.lstm_bias[512 + n];
      float bo = P.lstm_bias[768 + n];
#pragma unroll
      for (int r = 0; r < 8; ++r) {
        int m = mb + r;
        float gi = qa[0][r] + bi;
        float gf = qa[1][r] + bf;
        float gg = qa[2][r] + bg;
        float go = qa[3][r] + bo;
        float c  = s_c[m * 256 + n];
        float cn = sigmoidf_(gf) * c + sigmoidf_(gi) * tanh_(gg);
        s_c[m * 256 + n] = cn;           // exclusive per-lane ownership
        hn_keep[p][r] = sigmoidf_(go) * tanh_(cn);
      }
    }
    __syncthreads();   // all waves done reading s_hb; now safe to overwrite

    // commit h_next
    {
      int mb = (lane & 16) ? 8 : 0;
#pragma unroll
      for (int p = 0; p < 2; ++p) {
        int n = (w + 8 * p) * 16 + (lane & 15);
#pragma unroll
        for (int r = 0; r < 8; ++r)
          s_hb[(mb + r) * 256 + n] = f2bf(hn_keep[p][r]);
      }
    }

    // dec_mu / dec_sigma -> global (reads s_dh only; no conflict with h update)
    { v8f acc = {}; acc = mmA(s_dh, 128, P.dec_muT, 128, w * 16, 4, 0, 0, acc, lane);
      dstore_gmem(P.o_decmu, b0, t, w * 16, acc, P.dec_mu_b, false, lane); }
    { v8f acc = {}; acc = mmA(s_dh, 128, P.dec_sgT, 128, w * 16, 4, 0, 0, acc, lane);
      dstore_gmem(P.o_decsg, b0, t, w * 16, acc, P.dec_sigma_b, true, lane); }
    __syncthreads();
  }
}

extern "C" void kernel_launch(void* const* d_in, const int* in_sizes, int n_in,
                              void* d_out, int out_size, void* d_ws, size_t ws_size,
                              hipStream_t stream) {
  (void)in_sizes; (void)n_in; (void)out_size; (void)ws_size;

  bf16_t* ws = (bf16_t*)d_ws;
  size_t off = 0;
  auto take = [&](size_t n) { bf16_t* p = ws + off; off += n; return p; };
  bf16_t* prior_hT  = take(128 * 256);
  bf16_t* prior_muT = take(128 * 128);
  bf16_t* prior_sgT = take(128 * 128);
  bf16_t* phi_xT    = take(128 * 128);
  bf16_t* enc_hT    = take(128 * 384);
  bf16_t* enc_muT   = take(128 * 128);
  bf16_t* enc_sgT   = take(128 * 128);
  bf16_t* phi_zT    = take(128 * 128);
  bf16_t* dec_hT    = take(128 * 384);
  bf16_t* dec_muT   = take(128 * 128);
  bf16_t* dec_sgT   = take(128 * 128);
  bf16_t* lstm_kT   = take(1024 * 256);
  bf16_t* lstm_recT = take(1024 * 256);

  auto conv = [&](int i, bf16_t* dst, int K, int N) {
    int total  = K * N;
    int blocks = (total + 255) / 256;
    wconv_kernel<<<blocks, 256, 0, stream>>>((const float*)d_in[i], dst, K, N);
  };
  conv(2,  prior_hT,  256, 128);
  conv(4,  prior_muT, 128, 128);
  conv(6,  prior_sgT, 128, 128);
  conv(8,  phi_xT,    128, 128);
  conv(10, enc_hT,    384, 128);
  conv(12, enc_muT,   128, 128);
  conv(14, enc_sgT,   128, 128);
  conv(16, phi_zT,    128, 128);
  conv(18, dec_hT,    384, 128);
  conv(20, dec_muT,   128, 128);
  conv(22, dec_sgT,   128, 128);
  conv(24, lstm_kT,   256, 1024);
  conv(25, lstm_recT, 256, 1024);

  float* out = (float*)d_out;
  const size_t seq = (size_t)B_ * T_ * 128;

  VrnnParams P;
  P.x   = (const float*)d_in[0];
  P.eps = (const float*)d_in[1];
  P.prior_h_b     = (const float*)d_in[3];
  P.prior_mu_b    = (const float*)d_in[5];
  P.prior_sigma_b = (const float*)d_in[7];
  P.phi_x_b       = (const float*)d_in[9];
  P.enc_h_b       = (const float*)d_in[11];
  P.enc_mu_b      = (const float*)d_in[13];
  P.enc_sigma_b   = (const float*)d_in[15];
  P.phi_z_b       = (const float*)d_in[17];
  P.dec_h_b       = (const float*)d_in[19];
  P.dec_mu_b      = (const float*)d_in[21];
  P.dec_sigma_b   = (const float*)d_in[23];
  P.lstm_bias     = (const float*)d_in[26];
  P.phi_xT = phi_xT;   P.prior_hT = prior_hT; P.prior_muT = prior_muT; P.prior_sgT = prior_sgT;
  P.enc_hT = enc_hT;   P.enc_muT  = enc_muT;  P.enc_sgT   = enc_sgT;   P.phi_zT    = phi_zT;
  P.dec_hT = dec_hT;   P.dec_muT  = dec_muT;  P.dec_sgT   = dec_sgT;
  P.lstm_kT = lstm_kT; P.lstm_recT = lstm_recT;
  P.o_rnn   = out;
  P.o_encmu = out + (size_t)B_ * Z_;
  P.o_encsg = P.o_encmu + seq;
  P.o_decmu = P.o_encsg + seq;
  P.o_decsg = P.o_decmu + seq;
  P.o_primu = P.o_decsg + seq;
  P.o_prisg = P.o_primu + seq;

  vrnn_step_kernel<<<B_ / 16, 256, 0, stream>>>(P);
}
